// ALiBiAttention_2302102471031
// MI455X (gfx1250) — matrix operand order repaired and checked by reading, not compiled, not run on hardware
//
#include <hip/hip_runtime.h>
#include <hip/hip_bf16.h>
#include <stdint.h>

// ---------------------------------------------------------------------------
// ALiBi causal attention for MI455X (gfx1250), bf16 WMMA flash-attention.
// B=2, T=2048, D=1024, H=16, hd=64
// Register-blocked: 32x64 tiles/wave in the projection GEMM, 32-row query
// tiles/wave in attention (K/V fragments reused across two M-tiles).
// ---------------------------------------------------------------------------

#define BB 2
#define TT 2048
#define DD 1024
#define HH 16
#define HD 64

typedef __attribute__((ext_vector_type(16))) __bf16 v16bf;
typedef __attribute__((ext_vector_type(8)))  float  v8f;

union Frag {
    v16bf    h;     // 16 bf16 values (8 VGPRs, 2 packed per VGPR)
    uint32_t u[8];
    uint4    q[2];
};

__device__ __forceinline__ uint16_t f2bf(float f) {
    uint32_t u = __float_as_uint(f);
    uint32_t r = u + 0x7FFFu + ((u >> 16) & 1u);   // round-to-nearest-even
    return (uint16_t)(r >> 16);
}

__device__ __forceinline__ v8f wmma_bf16(const Frag& a, const Frag& b, v8f c) {
    return __builtin_amdgcn_wmma_f32_16x16x32_bf16(
        /*neg_a=*/false, a.h, /*neg_b=*/false, b.h,
        /*c_mod=*/(short)0, c, /*reuse_a=*/false, /*reuse_b=*/false);
}

// ---------------------------------------------------------------------------
// Kernel 1: f32 -> bf16 conversion
// ---------------------------------------------------------------------------
__global__ __launch_bounds__(256) void cvt_bf16_kernel(const float* __restrict__ src,
                                                       uint16_t* __restrict__ dst, int n) {
    int i = blockIdx.x * 256 + threadIdx.x;
    if (i < n) dst[i] = f2bf(src[i]);
}

// ---------------------------------------------------------------------------
// Kernel 2: v = einsum('btjd,ij->bitd'), stored TRANSPOSED as Vt[b,i,d,t] bf16
// so attention B-fragments (keys along K dim) are contiguous b128 loads.
// ---------------------------------------------------------------------------
__global__ __launch_bounds__(256) void vmix_kernel(const float* __restrict__ xt,
                                                   const float* __restrict__ v_fact,
                                                   uint16_t* __restrict__ Vt) {
    int idx = blockIdx.x * 256 + threadIdx.x;          // B*H*HD*T threads
    int t = idx & (TT - 1);
    int r = idx / TT;
    int d = r & (HD - 1); r >>= 6;
    int i = r & (HH - 1);
    int b = r >> 4;
    float acc = 0.f;
    const float* xrow = xt + ((size_t)(b * TT + t)) * DD + d;
#pragma unroll
    for (int j = 0; j < HH; ++j)
        acc = fmaf(v_fact[i * HH + j], xrow[j * HD], acc);
    Vt[((size_t)((b * HH + i) * HD + d)) * TT + t] = f2bf(acc);
}

// ---------------------------------------------------------------------------
// Kernel 3: qk = x_norm @ qk_w^T + qk_b ; scatter q,k to (B,H,T,hd) bf16.
// 32x64 output tile per wave: 2 A-frags x 4 B-frags -> 8 WMMA per K-step,
// 12 b128 loads per K-step (43 FLOP/byte vs 10.7 unblocked).
// ---------------------------------------------------------------------------
__global__ __launch_bounds__(256) void qk_gemm_kernel(const uint16_t* __restrict__ Xbf,
                                                      const uint16_t* __restrict__ Wbf,
                                                      const float* __restrict__ bias,
                                                      uint16_t* __restrict__ Qbf,
                                                      uint16_t* __restrict__ Kbf) {
    const int lane = threadIdx.x & 31;
    const int wid  = (blockIdx.x * 256 + threadIdx.x) >> 5;   // 4096 waves
    const int tm = wid >> 5;              // 128 row blocks (32 rows over B*T)
    const int tn = wid & 31;              // 32 col blocks (64 cols over 2*D)
    const int rowbase = tm << 5, colbase = tn << 6;
    const int nh = lane & 15, half = lane >> 4;

    const uint16_t* arow[2];
    const uint16_t* brow[4];
#pragma unroll
    for (int mi = 0; mi < 2; ++mi)
        arow[mi] = Xbf + (size_t)(rowbase + mi * 16 + nh) * DD;   // A: lane = M
#pragma unroll
    for (int ni = 0; ni < 4; ++ni)
        brow[ni] = Wbf + (size_t)(colbase + ni * 16 + nh) * DD;   // B: lane = N

    v8f acc[2][4] = {};
    for (int kk = 0; kk < DD; kk += 32) {
        Frag a[2], b[4];
#pragma unroll
        for (int mi = 0; mi < 2; ++mi) {
            // A 16x32 bf16: VGPR0-3 K = half*8+0..7, VGPR4-7 K = 16+half*8+0..7
            a[mi].q[0] = *(const uint4*)(arow[mi] + kk + half * 8);
            a[mi].q[1] = *(const uint4*)(arow[mi] + kk + 16 + half * 8);
        }
#pragma unroll
        for (int ni = 0; ni < 4; ++ni) {
            // B 32x16 bf16: VGPR0-3 K = half*8+0..7, VGPR4-7 K = 16+half*8+0..7
            b[ni].q[0] = *(const uint4*)(brow[ni] + kk + half * 8);
            b[ni].q[1] = *(const uint4*)(brow[ni] + kk + 16 + half * 8);
        }
#pragma unroll
        for (int mi = 0; mi < 2; ++mi)
#pragma unroll
            for (int ni = 0; ni < 4; ++ni)
                acc[mi][ni] = wmma_bf16(a[mi], b[ni], acc[mi][ni]);
    }

#pragma unroll
    for (int ni = 0; ni < 4; ++ni) {
        const int o  = colbase + ni * 16 + nh;    // output column in [0, 2048)
        const float bv = bias[o];
        uint16_t* dst = (o < DD) ? Qbf : Kbf;
        const int oc  = o & (DD - 1);
        const int head = oc >> 6, dcol = oc & (HD - 1);
#pragma unroll
        for (int mi = 0; mi < 2; ++mi)
#pragma unroll
            for (int v = 0; v < 8; ++v) {
                int r = rowbase + mi * 16 + v + 8 * half;  // C row = v + 8*(lane>>4)
                int b_ = r >> 11, t = r & (TT - 1);
                dst[((size_t)((b_ * HH + head) * TT + t)) * HD + dcol] =
                    f2bf(acc[mi][ni][v] + bv);
            }
    }
}

// ---------------------------------------------------------------------------
// Kernel 4: flash attention with ALiBi + causal. 32-row query tile per wave;
// K/V fragments loaded once per 32-key block feed both M-tiles (16 WMMA per
// 16 b128 loads -> 64 FLOP/byte).
// ---------------------------------------------------------------------------
__global__ __launch_bounds__(256) void attn_kernel(const uint16_t* __restrict__ Qbf,
                                                   const uint16_t* __restrict__ Kbf,
                                                   const uint16_t* __restrict__ Vt,
                                                   float* __restrict__ Yatt) {
    __shared__ __align__(16) uint16_t plds[8][32 * 32];   // per-wave P staging (2 KB)

    const int lane = threadIdx.x & 31;
    const int w    = threadIdx.x >> 5;
    const int wid  = blockIdx.x * 8 + w;
    const int bh   = wid >> 6;            // T/32 = 64 query tiles per (b,h)
    const int qt   = wid & 63;
    const int qbase = qt << 5;
    const int nh = lane & 15, half = lane >> 4;

    const float slope = exp2f(-0.5f * (float)((bh & (HH - 1)) + 1));  // ALiBi, H=16
    const float scale = 0.125f;                                        // 1/sqrt(64)

    // Q A-fragments: 2 M-tiles x 2 K-chunks (hd = 64), reused for all key blocks
    Frag qa[2][2];
#pragma unroll
    for (int mi = 0; mi < 2; ++mi) {
        const uint16_t* qrowp = Qbf + ((size_t)bh * TT + qbase + mi * 16 + nh) * HD;
#pragma unroll
        for (int i = 0; i < 2; ++i) {
            qa[mi][i].q[0] = *(const uint4*)(qrowp + i * 32 + half * 8);
            qa[mi][i].q[1] = *(const uint4*)(qrowp + i * 32 + 16 + half * 8);
        }
    }

    v8f o[2][4] = {};                      // 32 x 64 f32 output accumulator
    float mrow[2][8], lrow[2][8];
#pragma unroll
    for (int mi = 0; mi < 2; ++mi)
#pragma unroll
        for (int v = 0; v < 8; ++v) { mrow[mi][v] = -INFINITY; lrow[mi][v] = 0.f; }

    const int nkb = qt + 1;                // keys 0 .. qbase+31, exact 32-blocks
    for (int kb = 0; kb < nkb; ++kb) {
        const int kbase = kb << 5;

        // ---- K fragments: 2 key tiles x 2 hd-chunks, loaded once ----
        const uint16_t* k0p = Kbf + ((size_t)bh * TT + kbase + nh) * HD;
        __builtin_prefetch(k0p + 32 * HD, 0, 1);   // next key block
        Frag kf[2][2];
#pragma unroll
        for (int kt = 0; kt < 2; ++kt)
#pragma unroll
            for (int i = 0; i < 2; ++i) {
                const uint16_t* kp = k0p + kt * 16 * HD + i * 32 + half * 8;
                kf[kt][i].q[0] = *(const uint4*)(kp);
                kf[kt][i].q[1] = *(const uint4*)(kp + 16);
            }

        // ---- S = Q K^T : 2 M-tiles x 2 key tiles ----
        v8f s[2][2] = {};
#pragma unroll
        for (int mi = 0; mi < 2; ++mi)
#pragma unroll
            for (int kt = 0; kt < 2; ++kt)
#pragma unroll
                for (int i = 0; i < 2; ++i)
                    s[mi][kt] = wmma_bf16(qa[mi][i], kf[kt][i], s[mi][kt]);

        // ---- online softmax update ----
#pragma unroll
        for (int mi = 0; mi < 2; ++mi)
#pragma unroll
            for (int v = 0; v < 8; ++v) {
                const int m    = v + 8 * half;
                const int qrow = qbase + mi * 16 + m;
                const int key0 = kbase + nh, key1 = key0 + 16;
                float x0 = (key0 <= qrow)
                             ? fmaf(slope, (float)(key0 - qrow), s[mi][0][v] * scale)
                             : -INFINITY;
                float x1 = (key1 <= qrow)
                             ? fmaf(slope, (float)(key1 - qrow), s[mi][1][v] * scale)
                             : -INFINITY;
                float cand = fmaxf(x0, x1);
                cand = fmaxf(cand, __shfl_xor(cand, 1, 32));
                cand = fmaxf(cand, __shfl_xor(cand, 2, 32));
                cand = fmaxf(cand, __shfl_xor(cand, 4, 32));
                cand = fmaxf(cand, __shfl_xor(cand, 8, 32));
                const float mnew  = fmaxf(mrow[mi][v], cand);
                const float alpha = __expf(mrow[mi][v] - mnew);
                const float p0 = __expf(x0 - mnew);
                const float p1 = __expf(x1 - mnew);
                float psum = p0 + p1;
                psum += __shfl_xor(psum, 1, 32);
                psum += __shfl_xor(psum, 2, 32);
                psum += __shfl_xor(psum, 4, 32);
                psum += __shfl_xor(psum, 8, 32);
                lrow[mi][v] = lrow[mi][v] * alpha + psum;
                mrow[mi][v] = mnew;
                o[mi][0][v] *= alpha; o[mi][1][v] *= alpha;
                o[mi][2][v] *= alpha; o[mi][3][v] *= alpha;
                // stage P (bf16) row-major for the A-layout re-read
                plds[w][(mi * 16 + m) * 32 + nh]      = f2bf(p0);
                plds[w][(mi * 16 + m) * 32 + 16 + nh] = f2bf(p1);
            }

        // ---- P A-fragments from LDS (same-wave DS ops are in-order) ----
        Frag pa[2];
#pragma unroll
        for (int mi = 0; mi < 2; ++mi) {
            pa[mi].q[0] = *(const uint4*)&plds[w][(mi * 16 + nh) * 32 + half * 8];
            pa[mi].q[1] = *(const uint4*)&plds[w][(mi * 16 + nh) * 32 + 16 + half * 8];
        }

        // ---- V B-fragments loaded once, consumed by both M-tiles ----
        Frag vb[4];
#pragma unroll
        for (int f = 0; f < 4; ++f) {
            const uint16_t* vp = Vt + ((size_t)(bh * HD) + f * 16 + nh) * TT
                                    + kbase + half * 8;
            vb[f].q[0] = *(const uint4*)(vp);
            vb[f].q[1] = *(const uint4*)(vp + 16);
        }
#pragma unroll
        for (int mi = 0; mi < 2; ++mi)
#pragma unroll
            for (int f = 0; f < 4; ++f)
                o[mi][f] = wmma_bf16(pa[mi], vb[f], o[mi][f]);
    }

    // ---- normalize and write y_att (B,H,T,hd) f32 ----
#pragma unroll
    for (int mi = 0; mi < 2; ++mi)
#pragma unroll
        for (int v = 0; v < 8; ++v) {
            const float inv = 1.0f / lrow[mi][v];
            const int qrow = qbase + mi * 16 + v + 8 * half;
            float* yp = Yatt + ((size_t)bh * TT + qrow) * HD + nh;
            yp[0]  = o[mi][0][v] * inv;
            yp[16] = o[mi][1][v] * inv;
            yp[32] = o[mi][2][v] * inv;
            yp[48] = o[mi][3][v] * inv;
        }
}

// ---------------------------------------------------------------------------
// Kernel 5: out[b,t,i*64+d] = sum_j out_fact[i,j] * Yatt[b,j,t,d]
// ---------------------------------------------------------------------------
__global__ __launch_bounds__(256) void outmix_kernel(const float* __restrict__ Yatt,
                                                     const float* __restrict__ out_fact,
                                                     float* __restrict__ out) {
    int idx = blockIdx.x * 256 + threadIdx.x;    // B*T*D threads
    int col = idx & (DD - 1);
    int row = idx / DD;                          // in [0, B*T)
    int i = col >> 6, d = col & (HD - 1);
    int b = row >> 11, t = row & (TT - 1);
    float acc = 0.f;
#pragma unroll
    for (int j = 0; j < HH; ++j)
        acc = fmaf(out_fact[i * HH + j],
                   Yatt[((size_t)((b * HH + j) * TT + t)) * HD + d], acc);
    out[idx] = acc;
}

// ---------------------------------------------------------------------------
// Host-side launch
// ---------------------------------------------------------------------------
extern "C" void kernel_launch(void* const* d_in, const int* in_sizes, int n_in,
                              void* d_out, int out_size, void* d_ws, size_t ws_size,
                              hipStream_t stream) {
    const float* x_norm  = (const float*)d_in[0];
    const float* xt      = (const float*)d_in[1];
    const float* qk_w    = (const float*)d_in[2];
    const float* qk_b    = (const float*)d_in[3];
    const float* v_fact  = (const float*)d_in[4];
    const float* out_fact= (const float*)d_in[5];
    float* out = (float*)d_out;

    // workspace carve-up (bytes); total ~52 MB
    char* ws = (char*)d_ws;
    uint16_t* Xbf = (uint16_t*)(ws);                              // B*T*D bf16     = 8 MB
    uint16_t* Wbf = (uint16_t*)(ws + 8388608);                    // 2D*D bf16      = 4 MB
    uint16_t* Qbf = (uint16_t*)(ws + 12582912);                   // B*H*T*64 bf16  = 8 MB
    uint16_t* Kbf = (uint16_t*)(ws + 20971520);                   // + 2 KB tail pad
    uint16_t* Vt  = (uint16_t*)(ws + 29362176);                   // + 2 KB tail pad
    float*    Yatt= (float*)   (ws + 37752832);                   // B*H*T*64 f32   = 16 MB

    const int nX = BB * TT * DD;        // 4,194,304
    const int nW = 2 * DD * DD;         // 2,097,152

    cvt_bf16_kernel<<<nX / 256, 256, 0, stream>>>(x_norm, Xbf, nX);
    cvt_bf16_kernel<<<nW / 256, 256, 0, stream>>>(qk_w, Wbf, nW);
    vmix_kernel<<<(BB * HH * HD * TT) / 256, 256, 0, stream>>>(xt, v_fact, Vt);

    // 128 x 32 blocks of 32x64, 1 block/wave, 8 waves/CTA -> 512 CTAs
    qk_gemm_kernel<<<512, 256, 0, stream>>>(Xbf, Wbf, qk_b, Qbf, Kbf);

    // B*H*(T/32) = 2048 query-tile waves, 8 waves/CTA -> 256 CTAs
    attn_kernel<<<256, 256, 0, stream>>>(Qbf, Kbf, Vt, Yatt);

    outmix_kernel<<<nX / 256, 256, 0, stream>>>(Yatt, out_fact, out);
}
